// GraphAttentionalLayer_68075231642178
// MI455X (gfx1250) — matrix operand order repaired and checked by reading, not compiled, not run on hardware
//
#include <hip/hip_runtime.h>

// ---------------------------------------------------------------------------
// GAT layer, fused flash-attention style for gfx1250 (MI455X, wave32, WMMA)
//   N=8192, IN_F=128, HEADS=4, HID=64
// Pipeline:
//   k_cvt     : h -> bf16 (hb), W -> bf16 transposed (wt[n][k])
//   k_gemm    : Y = h@W via v_wmma_f32_16x16x32_bf16, stored transposed bf16
//               vt[(head*64+d)*8192 + n]   (V^T, contiguous in n for B-tiles)
//   k_srctgt  : src[h][n], tgt[h][n] dot products with attention vector a
//   k_tmax    : per-head max of tgt  ->  static softmax max bound
//               (scores are rank-1: max_j leaky(src_i+tgt_j) <= leaky(src_i+max tgt),
//                so no online rescaling is needed in the j sweep)
//   k_flash   : per (16-row block, head): one-pass softmax-accumulate over
//               j-tiles of 32; P-tile (bf16) x V-tile (bf16) -> f32 WMMA acc
//   k_mean    : mean over heads -> d_out
// ---------------------------------------------------------------------------

typedef __attribute__((ext_vector_type(16))) __bf16 v16bf;
typedef __attribute__((ext_vector_type(8)))  __bf16 v8bf;
typedef __attribute__((ext_vector_type(8)))  float  v8f;
typedef __attribute__((ext_vector_type(4)))  float  v4f;
typedef __attribute__((ext_vector_type(4)))  int    v4i;

union BF16x16 { v16bf v; v8bf h[2]; };

#define GAT_N     8192
#define GAT_INF   128
#define GAT_HEADS 4
#define GAT_HID   64

// ---------------------------------------------------------------- convert
__global__ __launch_bounds__(256) void k_cvt(const float* __restrict__ h,
                                             const float* __restrict__ W,
                                             __bf16* __restrict__ hb,
                                             __bf16* __restrict__ wt) {
  int t = blockIdx.x * 256 + threadIdx.x;
  if (t < GAT_N * GAT_INF) hb[t] = (__bf16)h[t];
  if (t < 256 * GAT_INF) {                 // wt[n][k] = W[k][n]
    int n = t >> 7, k = t & 127;
    wt[t] = (__bf16)W[k * 256 + n];
  }
}

// ------------------------------------------------- feature GEMM (Y = h @ W)
// one wave per 16x16 output tile; K=128 -> 4 WMMA steps of K=32.
__global__ __launch_bounds__(32) void k_gemm(const __bf16* __restrict__ hb,
                                             const __bf16* __restrict__ wt,
                                             __bf16* __restrict__ vt) {
  const int lane = threadIdx.x;
  const int row  = lane & 15;
  const int half = lane >> 4;
  const int i0 = blockIdx.x * 16;          // output row block (n index)
  const int n0 = blockIdx.y * 16;          // output col block (head*64+d)
  v8f acc = {};
#pragma unroll
  for (int k0 = 0; k0 < GAT_INF; k0 += 32) {
    BF16x16 a, b;
    const __bf16* ap = hb + (size_t)(i0 + row) * GAT_INF + k0 + half * 8;
    a.h[0] = *(const v8bf*)(ap);
    a.h[1] = *(const v8bf*)(ap + 16);
    const __bf16* bp = wt + (size_t)(n0 + row) * GAT_INF + k0 + half * 8;
    b.h[0] = *(const v8bf*)(bp);
    b.h[1] = *(const v8bf*)(bp + 16);
    acc = __builtin_amdgcn_wmma_f32_16x16x32_bf16(false, a.v, false, b.v,
                                                  (short)0, acc, false, false);
  }
  // C layout: VGPR r, lane l -> (M = r + 8*half, N = l%16)
  v8bf st;
#pragma unroll
  for (int r = 0; r < 8; ++r) st[r] = (__bf16)acc[r];
  *(v8bf*)(vt + (size_t)(n0 + row) * GAT_N + i0 + half * 8) = st;
}

// -------------------------------------------------------------- src / tgt
__global__ __launch_bounds__(256) void k_srctgt(const __bf16* __restrict__ vt,
                                                const float* __restrict__ a,
                                                float* __restrict__ src,
                                                float* __restrict__ tgt) {
  int t = blockIdx.x * 256 + threadIdx.x;       // (head, n)
  if (t >= GAT_HEADS * GAT_N) return;
  int h = t >> 13, n = t & (GAT_N - 1);
  float s = 0.f, g = 0.f;
#pragma unroll 8
  for (int d = 0; d < GAT_HID; ++d) {
    float v = (float)vt[(size_t)(h * GAT_HID + d) * GAT_N + n];
    s += v * a[h * 128 + d];
    g += v * a[h * 128 + GAT_HID + d];
  }
  src[t] = s;
  tgt[t] = g;
}

// ------------------------------------------------------- per-head tgt max
__global__ __launch_bounds__(256) void k_tmax(const float* __restrict__ tgt,
                                              float* __restrict__ tmax) {
  __shared__ float red[256];
  const float* t = tgt + blockIdx.x * GAT_N;
  float m = -1e30f;
  for (int i = threadIdx.x; i < GAT_N; i += 256) m = fmaxf(m, t[i]);
  red[threadIdx.x] = m;
  __syncthreads();
  for (int s = 128; s > 0; s >>= 1) {
    if (threadIdx.x < s)
      red[threadIdx.x] = fmaxf(red[threadIdx.x], red[threadIdx.x + s]);
    __syncthreads();
  }
  if (threadIdx.x == 0) tmax[blockIdx.x] = red[0];
}

// -------------------------------------------------- fused flash attention
// grid (N/16, HEADS), 1 wave/block. Static max bound -> no online rescale,
// no cross-lane ops in the j loop.
__global__ __launch_bounds__(32) void k_flash(const int* __restrict__ adj,
                                              const __bf16* __restrict__ vt,
                                              const float* __restrict__ src,
                                              const float* __restrict__ tgt,
                                              const float* __restrict__ tmax,
                                              float* __restrict__ hp) {
  const int lane = threadIdx.x;
  const int row  = lane & 15;
  const int half = lane >> 4;
  const int i0   = blockIdx.x * 16;
  const int head = blockIdx.y;

  const float  s_i   = src[head * GAT_N + i0 + row];
  const float* tg    = tgt + head * GAT_N;
  const int*   arow  = adj + (size_t)(i0 + row) * GAT_N;
  const __bf16* vbase = vt + (size_t)head * GAT_HID * GAT_N;

  const float LOG2E = 1.44269504f;
  // row-max upper bound (leaky is monotonic): m_i = leaky(src_i + max_j tgt_j)
  const float xm  = s_i + tmax[head];
  const float mb  = fmaxf(xm, 0.1f * xm) * LOG2E;   // in log2 domain

  v8f acc[4] = {{}, {}, {}, {}};
  float lsum = 0.f;                                  // per-lane partial row sum

  for (int jb = 0; jb < GAT_N; jb += 32) {
    // prefetch the streaming adjacency row ~8KB ahead (global_prefetch_b8)
    if (jb + 2048 < GAT_N) __builtin_prefetch(arow + jb + 2048, 0, 1);

    // A-tile element c -> j = jb + 8*half + c        (c<8)
    //                     j = jb + 16 + 8*half + c-8 (c>=8)
    const int jr0 = jb + 8 * half;
    const int jr1 = jb + 16 + 8 * half;
    v4i ad0 = *(const v4i*)(arow + jr0);
    v4i ad1 = *(const v4i*)(arow + jr0 + 4);
    v4i ad2 = *(const v4i*)(arow + jr1);
    v4i ad3 = *(const v4i*)(arow + jr1 + 4);
    v4f t0  = *(const v4f*)(tg + jr0);
    v4f t1  = *(const v4f*)(tg + jr0 + 4);
    v4f t2  = *(const v4f*)(tg + jr1);
    v4f t3  = *(const v4f*)(tg + jr1 + 4);

    BF16x16 p;
#pragma unroll
    for (int c = 0; c < 16; ++c) {
      float tj = (c < 4) ? t0[c & 3] : (c < 8) ? t1[c & 3]
               : (c < 12) ? t2[c & 3] : t3[c & 3];
      int   aj = (c < 4) ? ad0[c & 3] : (c < 8) ? ad1[c & 3]
               : (c < 12) ? ad2[c & 3] : ad3[c & 3];
      float x  = s_i + tj;
      x = fmaxf(x, 0.1f * x);                           // LeakyReLU
      float pv = __builtin_amdgcn_exp2f(__builtin_fmaf(x, LOG2E, -mb));
      pv = (aj == 0) ? 0.f : pv;                        // adjacency mask
      lsum += pv;
      p.v[c] = (__bf16)pv;
    }

    // P(16x32) x V(32x16 per 16-dim chunk), f32 accumulate
#pragma unroll
    for (int chunk = 0; chunk < 4; ++chunk) {
      const __bf16* vp = vbase + (size_t)(chunk * 16 + row) * GAT_N + jb + 8 * half;
      BF16x16 b;
      b.h[0] = *(const v8bf*)(vp);
      b.h[1] = *(const v8bf*)(vp + 16);
      acc[chunk] = __builtin_amdgcn_wmma_f32_16x16x32_bf16(
          false, p.v, false, b.v, (short)0, acc[chunk], false, false);
    }
  }

  // combine the two j half-tiles' partial sums: lanes r and r+16 share a row
  const float l   = lsum + __shfl_xor(lsum, 16, 32);
  const float inv = 1.0f / l;
#pragma unroll
  for (int r = 0; r < 8; ++r) {
    float wlo = __shfl(inv, r, 32);
    float whi = __shfl(inv, r + 8, 32);
    float w   = half ? whi : wlo;
    int orow  = i0 + r + 8 * half;
    float* op = hp + ((size_t)head * GAT_N + orow) * GAT_HID + row;
#pragma unroll
    for (int chunk = 0; chunk < 4; ++chunk)
      op[chunk * 16] = acc[chunk][r] * w;
  }
}

// --------------------------------------------------------- mean over heads
__global__ __launch_bounds__(256) void k_mean(const float* __restrict__ hp,
                                              float* __restrict__ out) {
  int t = blockIdx.x * 256 + threadIdx.x;
  if (t >= GAT_N * GAT_HID) return;
  const size_t S = (size_t)GAT_N * GAT_HID;
  out[t] = 0.25f * (hp[t] + hp[t + S] + hp[t + 2 * S] + hp[t + 3 * S]);
}

// ---------------------------------------------------------------- launcher
extern "C" void kernel_launch(void* const* d_in, const int* in_sizes, int n_in,
                              void* d_out, int out_size, void* d_ws, size_t ws_size,
                              hipStream_t stream) {
  const float* h   = (const float*)d_in[0];   // (8192,128) f32
  const int*   adj = (const int*)  d_in[1];   // (8192,8192) i32
  const float* W   = (const float*)d_in[2];   // (128,256) f32
  const float* a   = (const float*)d_in[3];   // (4,128,1) f32
  float* out = (float*)d_out;                 // (8192,64) f32

  char* ws = (char*)d_ws;
  // workspace layout (16B-aligned offsets)
  __bf16* hb   = (__bf16*)(ws);                        // 8192*128*2  = 2 MiB
  __bf16* wt   = (__bf16*)(ws + 2097152);              // 256*128*2   = 64 KiB
  __bf16* vt   = (__bf16*)(ws + 2162688);              // 4*64*8192*2 = 4 MiB
  float*  src  = (float*) (ws + 6356992);              // 4*8192*4
  float*  tgt  = (float*) (ws + 6488064);              // 4*8192*4
  float*  hp   = (float*) (ws + 6619136);              // 4*8192*64*4 = 8 MiB
  float*  tmax = (float*) (ws + 15007744);             // 4*4 B

  k_cvt   <<<dim3((GAT_N * GAT_INF + 255) / 256), 256, 0, stream>>>(h, W, hb, wt);
  k_gemm  <<<dim3(GAT_N / 16, (GAT_HEADS * GAT_HID) / 16), 32, 0, stream>>>(hb, wt, vt);
  k_srctgt<<<dim3((GAT_HEADS * GAT_N) / 256), 256, 0, stream>>>(vt, a, src, tgt);
  k_tmax  <<<dim3(GAT_HEADS), 256, 0, stream>>>(tgt, tmax);
  k_flash <<<dim3(GAT_N / 16, GAT_HEADS), 32, 0, stream>>>(adj, vt, src, tgt, tmax, hp);
  k_mean  <<<dim3((GAT_N * GAT_HID) / 256), 256, 0, stream>>>(hp, out);
}
